// PointnetFPModule_57793079935585
// MI455X (gfx1250) — matrix operand order repaired and checked by reading, not compiled, not run on hardware
//
#include <hip/hip_runtime.h>
#include <hip/hip_bf16.h>
#include <stdint.h>

// ---------------------------------------------------------------------------
// PointNet++ FP module for MI455X (gfx1250, wave32, WMMA bf16 path)
//   B=16, n=4096, m=1024, C1=256, C2=512, mlp=[256,256]
// Fused design: three_nn -> gather/concat(bf16 A-matrix) -> fused 2-layer
// WMMA MLP (H strip never leaves LDS).
// ---------------------------------------------------------------------------

typedef __attribute__((ext_vector_type(16))) __bf16 v16bf;
typedef __attribute__((ext_vector_type(8)))  float  v8f;

union ABFrag { uint4 q[2]; v16bf v; };
union Pack8  { __bf16 h[8]; uint4 q; };

#define NB    16
#define NPTS  4096
#define MPTS  1024
#define C1C   256
#define C2C   512
#define CIN   768     // C1+C2
#define CMID  256     // mlp[0]
#define COUT  256     // mlp[1]

// ---------------------------------------------------------------------------
// Kernel 1: brute-force 3-NN + normalized inverse-squared-distance weights.
// ---------------------------------------------------------------------------
__global__ void __launch_bounds__(256)
three_nn_kernel(const float* __restrict__ unknown,
                const float* __restrict__ known,
                int* __restrict__ idx_out,
                float* __restrict__ w_out)
{
    __shared__ float kx[MPTS], ky[MPTS], kz[MPTS];

    const int t    = threadIdx.x;
    const int b    = blockIdx.x >> 4;          // 16 tiles per batch
    const int tile = blockIdx.x & 15;

    for (int j = t; j < MPTS; j += 256) {
        const float* kp = known + ((size_t)b * MPTS + j) * 3;
        kx[j] = kp[0]; ky[j] = kp[1]; kz[j] = kp[2];
    }
    __syncthreads();

    const int i = tile * 256 + t;
    const float* up = unknown + ((size_t)b * NPTS + i) * 3;
    const float px = up[0], py = up[1], pz = up[2];

    float d0 = 3.4e38f, d1 = 3.4e38f, d2 = 3.4e38f;
    int   i0 = 0,       i1 = 0,       i2 = 0;

    #pragma unroll 4
    for (int j = 0; j < MPTS; ++j) {
        const float dx = px - kx[j];
        const float dy = py - ky[j];
        const float dz = pz - kz[j];
        const float d  = dx * dx + dy * dy + dz * dz;
        if (d < d0)      { d2 = d1; i2 = i1; d1 = d0; i1 = i0; d0 = d; i0 = j; }
        else if (d < d1) { d2 = d1; i2 = i1; d1 = d;  i1 = j; }
        else if (d < d2) { d2 = d;  i2 = j; }
    }

    const float r0 = 1.0f / (d0 + 1e-8f);
    const float r1 = 1.0f / (d1 + 1e-8f);
    const float r2 = 1.0f / (d2 + 1e-8f);
    const float rs = 1.0f / (r0 + r1 + r2);

    const size_t base = ((size_t)b * NPTS + i) * 3;
    idx_out[base + 0] = i0;  w_out[base + 0] = r0 * rs;
    idx_out[base + 1] = i1;  w_out[base + 1] = r1 * rs;
    idx_out[base + 2] = i2;  w_out[base + 2] = r2 * rs;
}

// ---------------------------------------------------------------------------
// Kernel 2: three_interpolate gather + concat, emit bf16 activation matrix
// X (B*n, 768) row-major == WMMA A-operand order. 8 channels packed per
// uint4 so the row is written with global_store_b128.
// ---------------------------------------------------------------------------
__global__ void __launch_bounds__(256)
interp_concat_kernel(const float* __restrict__ known_feats,   // (B, C2, m)
                     const float* __restrict__ unknow_feats,  // (B, C1, n)
                     const int*   __restrict__ idx_in,
                     const float* __restrict__ w_in,
                     __bf16* __restrict__ X)                  // (B*n, 768)
{
    const int gid = blockIdx.x * 256 + threadIdx.x;   // 0 .. B*n-1
    const int b   = gid >> 12;
    const int i   = gid & (NPTS - 1);

    const size_t base = (size_t)gid * 3;
    const int   j0 = idx_in[base + 0], j1 = idx_in[base + 1], j2 = idx_in[base + 2];
    const float w0 = w_in[base + 0],  w1 = w_in[base + 1],  w2 = w_in[base + 2];

    const float* kf = known_feats + (size_t)b * C2C * MPTS;
    uint4* xr4 = (uint4*)(X + (size_t)gid * CIN);

    for (int c0 = 0; c0 < C2C; c0 += 8) {
        Pack8 p;
        #pragma unroll
        for (int u = 0; u < 8; ++u) {
            const float* row = kf + (size_t)(c0 + u) * MPTS;
            p.h[u] = (__bf16)(w0 * row[j0] + w1 * row[j1] + w2 * row[j2]);
        }
        xr4[c0 >> 3] = p.q;
    }

    const float* uf = unknow_feats + (size_t)b * C1C * NPTS + i;
    for (int c0 = 0; c0 < C1C; c0 += 8) {
        Pack8 p;
        #pragma unroll
        for (int u = 0; u < 8; ++u)
            p.h[u] = (__bf16)uf[(size_t)(c0 + u) * NPTS];
        xr4[(C2C + c0) >> 3] = p.q;
    }
}

// ---------------------------------------------------------------------------
// Kernel 3: convert W (K x 256, f32 row-major) to bf16 pre-swizzled into the
// WMMA B-operand lane layout, tiled [ktile][ntile][lane][16 halfwords]:
//   lane l -> column n = 16*ntile + (l%16); halfword h -> K = 32*kt + 16*(l/16) + h
// ---------------------------------------------------------------------------
__global__ void __launch_bounds__(256)
pack_w_kernel(const float* __restrict__ W, __bf16* __restrict__ Wp, int ktiles)
{
    const int id = blockIdx.x * 256 + threadIdx.x;
    if (id >= ktiles * 16 * 32 * 16) return;
    const int h  = id & 15;
    const int l  = (id >> 4) & 31;
    const int j  = (id >> 9) & 15;
    const int kt = id >> 13;
    const int k  = 32 * kt + 16 * (l >> 4) + h;
    const int n  = 16 * j + (l & 15);
    Wp[id] = (__bf16)W[(size_t)k * 256 + n];
}

// ---------------------------------------------------------------------------
// Helper: one 16x256 WMMA strip pass over the LDS-resident W slab, with
// software-pipelined B-fragment ds_loads. The A operand is identical across
// all 16 WMMAs of the strip, so iterations 1..15 carry the reuse_a hint
// (peeled first iteration keeps the modifier a literal constant, as the
// builtin requires).
// ---------------------------------------------------------------------------
__device__ __forceinline__ void
wmma_strip(const ABFrag& a, const uint4* Bs, int lane, v8f acc[16])
{
    ABFrag bcur, bnext;
    bcur.q[0] = Bs[lane * 2];
    bcur.q[1] = Bs[lane * 2 + 1];

    // ntl = 0: first of the identical WMMA sequence, no reuse hint.
    bnext.q[0] = Bs[64 + lane * 2];
    bnext.q[1] = Bs[64 + lane * 2 + 1];
    acc[0] = __builtin_amdgcn_wmma_f32_16x16x32_bf16(
        false, a.v, false, bcur.v, (short)0, acc[0], false, false);
    bcur = bnext;

    #pragma unroll
    for (int ntl = 1; ntl < 16; ++ntl) {
        if (ntl < 15) {
            bnext.q[0] = Bs[(ntl + 1) * 64 + lane * 2];
            bnext.q[1] = Bs[(ntl + 1) * 64 + lane * 2 + 1];
        }
        acc[ntl] = __builtin_amdgcn_wmma_f32_16x16x32_bf16(
            false, a.v, false, bcur.v, (short)0, acc[ntl],
            /*reuse_a=*/true, /*reuse_b=*/false);
        if (ntl < 15) bcur = bnext;
    }
}

// ---------------------------------------------------------------------------
// Kernel 4: fused 2-layer MLP.
//   Block = 128 threads (4 waves); each wave owns a 16-point strip.
//   Phase 1: H = relu(X(16x768) * W0(768x256))  -> bf16 strip kept in LDS.
//   Phase 2: out = relu(H(16x256) * W1(256x256)) -> f32 (B, 256, n).
// LDS: 16 KB W-slab staging + 4 x 8 KB per-wave H strips = 48 KB/block.
// H never touches global memory (saves ~67 MB of HBM traffic).
// ---------------------------------------------------------------------------
__global__ void __launch_bounds__(128)
fused_mlp_kernel(const __bf16* __restrict__ X,      // (B*n, 768)
                 const __bf16* __restrict__ W0p,    // packed, 24 slabs x 16 KB
                 const __bf16* __restrict__ W1p,    // packed,  8 slabs x 16 KB
                 float* __restrict__ out)           // (B, 256, n)
{
    __shared__ uint4  Bs[1024];             // 16 KB: one K-slab (16 ntiles x 1 KB)
    __shared__ __bf16 Hs[4 * 16 * CMID];    // 32 KB: per-wave 16x256 bf16 strips

    const int t       = threadIdx.x;
    const int lane    = t & 31;
    const int wave    = t >> 5;
    const int b       = blockIdx.x >> 6;        // 64 row-blocks per batch
    const int rowBlk  = (blockIdx.x & 63) * 64;
    const int waveRow = rowBlk + wave * 16;

    const int mrow = lane & 15;                 // A row within tile
    const int hi16 = lane >> 4;                 // lane half selector
    const int mbase = hi16 * 8;

    const uint4* Arow = (const uint4*)(X + ((size_t)b * NPTS + waveRow + mrow) * CIN);

    const v8f vz = {0.f,0.f,0.f,0.f,0.f,0.f,0.f,0.f};

    // ---------------- Phase 1: X * W0 ----------------
    v8f acc[16];
    #pragma unroll
    for (int ntl = 0; ntl < 16; ++ntl) acc[ntl] = vz;

    for (int kt = 0; kt < 24; ++kt) {
        __syncthreads();
        const uint4* slab = (const uint4*)W0p + (size_t)kt * 1024;
        #pragma unroll
        for (int r = 0; r < 8; ++r)
            Bs[t + 128 * r] = slab[t + 128 * r];
        __syncthreads();

        if (kt < 23)
            __builtin_prefetch(&Arow[(kt + 1) * 4 + hi16], 0, 0);

        ABFrag a;
        a.q[0] = Arow[kt * 4 + 2 * hi16];       // K = 32kt + 16*(l/16) .. +7
        a.q[1] = Arow[kt * 4 + 2 * hi16 + 1];   // K = 32kt + 16*(l/16) + 8 .. +15

        wmma_strip(a, Bs, lane, acc);
    }

    // ReLU + bf16, spill the wave's 16x256 H strip to its private LDS region.
    __bf16* hs = &Hs[wave * 16 * CMID];
    #pragma unroll
    for (int ntl = 0; ntl < 16; ++ntl) {
        const int n = ntl * 16 + (lane & 15);
        #pragma unroll
        for (int r = 0; r < 8; ++r) {
            float v = acc[ntl][r];
            v = v > 0.f ? v : 0.f;
            hs[(mbase + r) * CMID + n] = (__bf16)v;
        }
    }
    __syncthreads();   // Hs visible; Bs safe to overwrite

    // ---------------- Phase 2: H * W1 ----------------
    v8f acc2[16];
    #pragma unroll
    for (int ntl = 0; ntl < 16; ++ntl) acc2[ntl] = vz;

    const uint4* hrow = (const uint4*)(hs + (size_t)mrow * CMID);

    for (int kt = 0; kt < 8; ++kt) {
        __syncthreads();
        const uint4* slab = (const uint4*)W1p + (size_t)kt * 1024;
        #pragma unroll
        for (int r = 0; r < 8; ++r)
            Bs[t + 128 * r] = slab[t + 128 * r];
        __syncthreads();

        ABFrag a;
        a.q[0] = hrow[kt * 4 + 2 * hi16];
        a.q[1] = hrow[kt * 4 + 2 * hi16 + 1];

        wmma_strip(a, Bs, lane, acc2);
    }

    // ReLU + f32 store: D rows (M = r + 8*(l/16)) are consecutive points ->
    // two contiguous float4 stores per tile into (B, 256, n).
    float* ob = out + (size_t)b * COUT * NPTS;
    const int i0 = waveRow + mbase;
    #pragma unroll
    for (int ntl = 0; ntl < 16; ++ntl) {
        const int d = ntl * 16 + (lane & 15);
        float4 lo, hh;
        lo.x = fmaxf(acc2[ntl][0], 0.f);
        lo.y = fmaxf(acc2[ntl][1], 0.f);
        lo.z = fmaxf(acc2[ntl][2], 0.f);
        lo.w = fmaxf(acc2[ntl][3], 0.f);
        hh.x = fmaxf(acc2[ntl][4], 0.f);
        hh.y = fmaxf(acc2[ntl][5], 0.f);
        hh.z = fmaxf(acc2[ntl][6], 0.f);
        hh.w = fmaxf(acc2[ntl][7], 0.f);
        float* dst = ob + (size_t)d * NPTS + i0;
        *(float4*)(dst)     = lo;
        *(float4*)(dst + 4) = hh;
    }
}

// ---------------------------------------------------------------------------
// Workspace layout (bytes)
// ---------------------------------------------------------------------------
static const size_t OFF_IDX = 0;                         //  786,432  int
static const size_t OFF_W   = 786432;                    //  786,432  float
static const size_t OFF_W0P = 1572864;                   //  393,216  bf16
static const size_t OFF_W1P = 1966080;                   //  131,072  bf16
static const size_t OFF_X   = 2097152;                   // 100.7 MB  bf16
                                                         // total ~102.8 MB

extern "C" void kernel_launch(void* const* d_in, const int* in_sizes, int n_in,
                              void* d_out, int out_size, void* d_ws, size_t ws_size,
                              hipStream_t stream)
{
    const float* unknown      = (const float*)d_in[0];   // (B, n, 3)
    const float* known        = (const float*)d_in[1];   // (B, m, 3)
    const float* unknow_feats = (const float*)d_in[2];   // (B, C1, n)
    const float* known_feats  = (const float*)d_in[3];   // (B, C2, m)
    const float* W0           = (const float*)d_in[4];   // (768, 256)
    const float* W1           = (const float*)d_in[5];   // (256, 256)
    float* out = (float*)d_out;                          // (B, 256, n)

    char* ws = (char*)d_ws;
    int*    idx = (int*)   (ws + OFF_IDX);
    float*  wgt = (float*) (ws + OFF_W);
    __bf16* W0p = (__bf16*)(ws + OFF_W0P);
    __bf16* W1p = (__bf16*)(ws + OFF_W1P);
    __bf16* X   = (__bf16*)(ws + OFF_X);

    // 1) three_nn + weights
    three_nn_kernel<<<NB * (NPTS / 256), 256, 0, stream>>>(unknown, known, idx, wgt);

    // 2) weight repack (f32 -> bf16 WMMA-B layout)
    pack_w_kernel<<<(24 * 8192) / 256, 256, 0, stream>>>(W0, W0p, 24);
    pack_w_kernel<<<( 8 * 8192) / 256, 256, 0, stream>>>(W1, W1p, 8);

    // 3) gather-interpolate + concat -> bf16 A matrix
    interp_concat_kernel<<<(NB * NPTS) / 256, 256, 0, stream>>>(
        known_feats, unknow_feats, idx, wgt, X);

    // 4) fused 2-layer WMMA MLP (H stays in LDS)
    fused_mlp_kernel<<<NB * (NPTS / 64), 128, 0, stream>>>(X, W0p, W1p, out);
}
